// RNNAgent_old_26912265076969
// MI455X (gfx1250) — matrix operand order repaired and checked by reading, not compiled, not run on hardware
//
#include <hip/hip_runtime.h>

typedef __attribute__((ext_vector_type(16))) _Float16 v16h;
typedef __attribute__((ext_vector_type(8)))  float    v8f;

#define B_TOT 262144
#define IN_F  50
#define ROWS  256   // batch rows per block (8 waves x 32 rows)
#define XSTR  72    // LDS row stride (f16 elems): 144B, 16B-aligned, bank-skewed

union FragH { v16h v; uint4 q[2]; };

// packed gate row -> original W_ih row. Tiles: i0-15, i16-19, g0-15, g16-19, o0-15, o16-19
__device__ __forceinline__ int packRow(int p) {
  int t = p >> 4, c = p & 15;
  switch (t) {
    case 0:  return c;                       // i units 0-15   (rows 0-19 of W)
    case 1:  return (c < 4) ? 16 + c : -1;   // i units 16-19
    case 2:  return 40 + c;                  // g units 0-15   (rows 40-59)
    case 3:  return (c < 4) ? 56 + c : -1;   // g units 16-19
    case 4:  return 60 + c;                  // o units 0-15   (rows 60-79)
    default: return (c < 4) ? 76 + c : -1;   // o units 16-19
  }
}

// hardware tanh if the builtin exists (gfx1250 trans unit has V_TANH);
// else exp + fast rcp (v_exp_f32 + v_rcp_f32, no IEEE div expansion)
__device__ __forceinline__ float tanh_fast(float x) {
#if __has_builtin(__builtin_amdgcn_tanhf)
  return __builtin_amdgcn_tanhf(x);
#else
  float cx = fminf(fmaxf(x, -15.0f), 15.0f);
  float e  = __expf(2.0f * cx);
  return (e - 1.0f) * __builtin_amdgcn_rcpf(e + 1.0f);
#endif
}
__device__ __forceinline__ float sigm(float x) {
#if __has_builtin(__builtin_amdgcn_tanhf)
  return fmaf(0.5f, __builtin_amdgcn_tanhf(0.5f * x), 0.5f);
#else
  return __builtin_amdgcn_rcpf(1.0f + __expf(-x));
#endif
}
__device__ __forceinline__ v8f bcast8(float x) {
  v8f r;
#pragma unroll
  for (int e = 0; e < 8; ++e) r[e] = x;
  return r;
}

#define WMMA_F16(A, Bv, C) \
  __builtin_amdgcn_wmma_f32_16x16x32_f16(false, (A), false, (Bv), (short)0, (C), false, false)

__global__ __launch_bounds__(256) void k_fused(
    const float* __restrict__ inp, const float* __restrict__ wvec,
    const float* __restrict__ Wih, const float* __restrict__ bih,
    const float* __restrict__ bhh, const float* __restrict__ c1w,
    const float* __restrict__ c1b, const float* __restrict__ c2w,
    const float* __restrict__ c2b, float* __restrict__ logits,
    float* __restrict__ bmax)
{
  __shared__ __align__(16) _Float16 Wl[96][XSTR];    // packed LSTM weights (B-matrix rows)
  __shared__ __align__(16) _Float16 C1[32][XSTR];    // conv1 weights
  __shared__ __align__(16) _Float16 xb[ROWS][XSTR];  // input tile (per slice)
  __shared__ __align__(16) _Float16 hb[ROWS][XSTR];  // 60-channel hidden concat
  __shared__ float biasPk[96];
  __shared__ float sc[48];   // [0..20]=conv2_w, [21]=conv2_b, [22..41]=conv1_b
  __shared__ float wred[8];

  const int tid  = threadIdx.x;
  const int lane = tid & 31, wave = tid >> 5;
  const int half = lane >> 4, lc = lane & 15;
  const int rowBase = blockIdx.x * ROWS;

  // ---- stage weights / biases into LDS (f16, packed, zero-padded) ----
  for (int idx = tid; idx < 96 * 64; idx += 256) {
    int p = idx >> 6, k = idx & 63;
    int ro = packRow(p);
    float v = (ro >= 0 && k < IN_F) ? Wih[ro * IN_F + k] : 0.0f;
    Wl[p][k] = (_Float16)v;
  }
  for (int idx = tid; idx < 32 * 64; idx += 256) {
    int p = idx >> 6, k = idx & 63;
    float v = (p < 20 && k < 60) ? c1w[p * 60 + k] : 0.0f;
    C1[p][k] = (_Float16)v;
  }
  if (tid < 96) { int ro = packRow(tid); biasPk[tid] = (ro >= 0) ? bih[ro] + bhh[ro] : 0.0f; }
  if (tid < 21)  sc[tid] = c2w[tid];
  if (tid == 21) sc[21]  = c2b[0];
  if (tid >= 22 && tid < 42) sc[tid] = c1b[tid - 22];
  { // zero K-padding once; only k < valid is written afterwards
    int r = tid;
    for (int k = IN_F; k < 64; ++k) xb[r][k] = (_Float16)0.0f;
    for (int k = 60;   k < 64; ++k) hb[r][k] = (_Float16)0.0f;
  }
  __syncthreads();

  const float bI0 = biasPk[lc],      bI1 = biasPk[16 + lc];
  const float bG0 = biasPk[32 + lc], bG1 = biasPk[48 + lc];
  const float bO0 = biasPk[64 + lc], bO1 = biasPk[80 + lc];

  // ---- 3 LSTM slices: GEMM via WMMA + lane-local gate nonlinearity ----
  for (int s = 0; s < 3; ++s) {
    __syncthreads();  // everyone done reading previous xb
    const float* src = inp + ((size_t)s * B_TOT + rowBase) * IN_F;
    for (int idx = tid; idx < ROWS * IN_F; idx += 256) {
      int r = idx / IN_F, k = idx - r * IN_F;
      xb[r][k] = (_Float16)src[idx];
    }
    __syncthreads();

#pragma unroll
    for (int mt = 0; mt < 2; ++mt) {
      const int mbase = wave * 32 + mt * 16;
      v8f aI0 = bcast8(bI0), aI1 = bcast8(bI1);
      v8f aG0 = bcast8(bG0), aG1 = bcast8(bG1);
      v8f aO0 = bcast8(bO0), aO1 = bcast8(bO1);
#pragma unroll
      for (int kt = 0; kt < 2; ++kt) {
        FragH A;
        const int aRow = mbase + lc, k0 = kt * 32 + 16 * half;
        A.q[0] = *(const uint4*)&xb[aRow][k0];        // K = k0..k0+7
        A.q[1] = *(const uint4*)&xb[aRow][k0 + 8];    // K = k0+8..k0+15
        const int kb = kt * 32 + 16 * half;           // B frag: 16 contiguous K
        FragH Bf;
        Bf.q[0] = *(const uint4*)&Wl[lc][kb];      Bf.q[1] = *(const uint4*)&Wl[lc][kb + 8];
        aI0 = WMMA_F16(A.v, Bf.v, aI0);
        Bf.q[0] = *(const uint4*)&Wl[16 + lc][kb]; Bf.q[1] = *(const uint4*)&Wl[16 + lc][kb + 8];
        aI1 = WMMA_F16(A.v, Bf.v, aI1);
        Bf.q[0] = *(const uint4*)&Wl[32 + lc][kb]; Bf.q[1] = *(const uint4*)&Wl[32 + lc][kb + 8];
        aG0 = WMMA_F16(A.v, Bf.v, aG0);
        Bf.q[0] = *(const uint4*)&Wl[48 + lc][kb]; Bf.q[1] = *(const uint4*)&Wl[48 + lc][kb + 8];
        aG1 = WMMA_F16(A.v, Bf.v, aG1);
        Bf.q[0] = *(const uint4*)&Wl[64 + lc][kb]; Bf.q[1] = *(const uint4*)&Wl[64 + lc][kb + 8];
        aO0 = WMMA_F16(A.v, Bf.v, aO0);
        Bf.q[0] = *(const uint4*)&Wl[80 + lc][kb]; Bf.q[1] = *(const uint4*)&Wl[80 + lc][kb + 8];
        aO1 = WMMA_F16(A.v, Bf.v, aO1);
      }
      // h = sigmoid(o) * tanh( sigmoid(i) * tanh(g) )  (c0 = 0, f dead)
#pragma unroll
      for (int v = 0; v < 8; ++v) {
        const int row = mbase + v + 8 * half;
        float h0 = sigm(aO0[v]) * tanh_fast(sigm(aI0[v]) * tanh_fast(aG0[v]));
        hb[row][s * 20 + lc] = (_Float16)h0;
        if (lc < 4) {
          float h1 = sigm(aO1[v]) * tanh_fast(sigm(aI1[v]) * tanh_fast(aG1[v]));
          hb[row][s * 20 + 16 + lc] = (_Float16)h1;
        }
      }
    }
  }

  // ---- conv1 (WMMA) + conv2 (shfl reduce) -> logits; per-wave hb rows only ----
  const float w2a  = sc[lc];
  const float w2b  = (lc < 4) ? sc[16 + lc] : 0.0f;
  const float cb0  = sc[22 + lc];
  const float cb1  = (lc < 4) ? sc[38 + lc] : 0.0f;
  const float w2w  = sc[20], c2bb = sc[21];
  float lmax = -3.4e38f;

#pragma unroll
  for (int mt = 0; mt < 2; ++mt) {
    const int mbase = wave * 32 + mt * 16;
    v8f c0 = bcast8(cb0), c1a = bcast8(cb1);
#pragma unroll
    for (int kt = 0; kt < 2; ++kt) {
      FragH A;
      const int aRow = mbase + lc, k0 = kt * 32 + 16 * half;
      A.q[0] = *(const uint4*)&hb[aRow][k0];
      A.q[1] = *(const uint4*)&hb[aRow][k0 + 8];
      const int kb = kt * 32 + 16 * half;
      FragH Bf;
      Bf.q[0] = *(const uint4*)&C1[lc][kb];      Bf.q[1] = *(const uint4*)&C1[lc][kb + 8];
      c0  = WMMA_F16(A.v, Bf.v, c0);
      Bf.q[0] = *(const uint4*)&C1[16 + lc][kb]; Bf.q[1] = *(const uint4*)&C1[16 + lc][kb + 8];
      c1a = WMMA_F16(A.v, Bf.v, c1a);
    }
#pragma unroll
    for (int v = 0; v < 8; ++v) {
      float p = w2a * fmaxf(c0[v], 0.0f) + w2b * fmaxf(c1a[v], 0.0f);
      p += __shfl_xor(p, 1); p += __shfl_xor(p, 2);
      p += __shfl_xor(p, 4); p += __shfl_xor(p, 8);     // sum over 16 lanes = 20 channels
      const int grow = rowBase + mbase + v + 8 * half;
      float y = fmaxf(p + w2w * wvec[1 + grow] + c2bb, 0.0f);
      lmax = fmaxf(lmax, y);
      if (lc == 0) logits[grow] = y;
    }
  }
  lmax = fmaxf(lmax, __shfl_xor(lmax, 1));
  lmax = fmaxf(lmax, __shfl_xor(lmax, 2));
  lmax = fmaxf(lmax, __shfl_xor(lmax, 4));
  lmax = fmaxf(lmax, __shfl_xor(lmax, 8));
  lmax = fmaxf(lmax, __shfl_xor(lmax, 16));
  if (lane == 0) wred[wave] = lmax;
  __syncthreads();
  if (tid == 0) {
    float m = wred[0];
    for (int i = 1; i < 8; ++i) m = fmaxf(m, wred[i]);
    bmax[blockIdx.x] = m;
  }
}

// ---- softmax: deterministic tree reductions, no float atomics ----
__global__ __launch_bounds__(256) void k_maxred(const float* __restrict__ bmax,
                                                float* __restrict__ scal) {
  __shared__ float red[8];
  int tid = threadIdx.x, lane = tid & 31, wave = tid >> 5;
  float m = 1.0f;  // the appended trailing 1.0 element
  for (int i = tid; i < B_TOT / ROWS; i += 256) m = fmaxf(m, bmax[i]);
  m = fmaxf(m, __shfl_xor(m, 1));  m = fmaxf(m, __shfl_xor(m, 2));
  m = fmaxf(m, __shfl_xor(m, 4));  m = fmaxf(m, __shfl_xor(m, 8));
  m = fmaxf(m, __shfl_xor(m, 16));
  if (lane == 0) red[wave] = m;
  __syncthreads();
  if (tid == 0) {
    float mm = red[0];
    for (int i = 1; i < 8; ++i) mm = fmaxf(mm, red[i]);
    scal[0] = mm;
  }
}

__global__ __launch_bounds__(256) void k_expsum(const float* __restrict__ logits,
                                                const float* __restrict__ scal,
                                                float* __restrict__ out,
                                                float* __restrict__ bsum) {
  __shared__ float red[8];
  int tid = threadIdx.x, lane = tid & 31, wave = tid >> 5;
  int i = blockIdx.x * 256 + tid;
  float e = __expf(logits[i] - scal[0]);
  out[i] = e;
  float s = e;
  s += __shfl_xor(s, 1); s += __shfl_xor(s, 2);
  s += __shfl_xor(s, 4); s += __shfl_xor(s, 8); s += __shfl_xor(s, 16);
  if (lane == 0) red[wave] = s;
  __syncthreads();
  if (tid == 0) {
    float t = red[0];
    for (int k = 1; k < 8; ++k) t += red[k];
    bsum[blockIdx.x] = t;
  }
}

__global__ __launch_bounds__(256) void k_sumred(const float* __restrict__ bsum,
                                                float* __restrict__ scal,
                                                float* __restrict__ out) {
  __shared__ float red[8];
  int tid = threadIdx.x, lane = tid & 31, wave = tid >> 5;
  float s = 0.0f;
  for (int i = tid; i < B_TOT / 256; i += 256) s += bsum[i];
  s += __shfl_xor(s, 1); s += __shfl_xor(s, 2);
  s += __shfl_xor(s, 4); s += __shfl_xor(s, 8); s += __shfl_xor(s, 16);
  if (lane == 0) red[wave] = s;
  __syncthreads();
  if (tid == 0) {
    float t = red[0];
    for (int k = 1; k < 8; ++k) t += red[k];
    float eB  = __expf(1.0f - scal[0]);
    float inv = 1.0f / (t + eB);
    scal[1] = inv;
    out[B_TOT] = eB * inv;
  }
}

__global__ __launch_bounds__(256) void k_scale(float* __restrict__ out,
                                               const float* __restrict__ scal) {
  int i = blockIdx.x * 256 + threadIdx.x;
  out[i] *= scal[1];
}

extern "C" void kernel_launch(void* const* d_in, const int* in_sizes, int n_in,
                              void* d_out, int out_size, void* d_ws, size_t ws_size,
                              hipStream_t stream) {
  (void)in_sizes; (void)n_in; (void)out_size; (void)ws_size;
  const float* inp = (const float*)d_in[0];
  const float* wv  = (const float*)d_in[1];
  const float* Wih = (const float*)d_in[2];
  // d_in[3] = W_hh: provably unused (single step, h0 = 0)
  const float* bih = (const float*)d_in[4];
  const float* bhh = (const float*)d_in[5];
  const float* c1w = (const float*)d_in[6];
  const float* c1b = (const float*)d_in[7];
  const float* c2w = (const float*)d_in[8];
  const float* c2b = (const float*)d_in[9];
  float* out = (float*)d_out;
  float* wsf = (float*)d_ws;
  float* logits = wsf;                    // B floats
  float* bmax   = wsf + B_TOT;            // 1024 floats
  float* bsum   = wsf + B_TOT + 1024;     // 1024 floats
  float* scal   = wsf + B_TOT + 2048;     // [0]=max, [1]=1/sum

  k_fused <<<B_TOT / ROWS, 256, 0, stream>>>(inp, wv, Wih, bih, bhh, c1w, c1b,
                                             c2w, c2b, logits, bmax);
  k_maxred<<<1,            256, 0, stream>>>(bmax, scal);
  k_expsum<<<B_TOT / 256,  256, 0, stream>>>(logits, scal, out, bsum);
  k_sumred<<<1,            256, 0, stream>>>(bsum, scal, out);
  k_scale <<<B_TOT / 256,  256, 0, stream>>>(out, scal);
}
